// QueryGuidedMoELinear_44839458570558
// MI455X (gfx1250) — matrix operand order repaired and checked by reading, not compiled, not run on hardware
//
#include <hip/hip_runtime.h>

// ---- CDNA5 (gfx1250, wave32) types ----
typedef __attribute__((ext_vector_type(16))) __bf16 bf16x16;
typedef __attribute__((ext_vector_type(8)))  __bf16 bf16x8;
typedef __attribute__((ext_vector_type(4)))  __bf16 bf16x4;
typedef __attribute__((ext_vector_type(8)))  float  f32x8;

#define B_   32
#define L_   1024
#define IN_  1024
#define OUT_ 512
#define NE_  8

// ---------------------------------------------------------------------------
// Kernel 1: mix expert weights with gates -> per-batch bf16 weight [B][OUT][IN]
// ---------------------------------------------------------------------------
__global__ __launch_bounds__(256) void moe_mix_w(const float* __restrict__ gates,
                                                 const float* __restrict__ weight,
                                                 __bf16* __restrict__ wbf) {
  const unsigned tid  = blockIdx.x * 256u + threadIdx.x;     // B*OUT*IN/4 threads
  const unsigned perB = (OUT_ * IN_) / 4;                    // 131072
  const unsigned b    = tid / perB;
  const unsigned r4   = (tid % perB) * 4;                    // o*IN + i, i%4==0

  float g[NE_];
#pragma unroll
  for (int e = 0; e < NE_; ++e) g[e] = gates[b * NE_ + e];

  float ax = 0.f, ay = 0.f, az = 0.f, aw = 0.f;
#pragma unroll
  for (int e = 0; e < NE_; ++e) {
    const float4 wv = *(const float4*)(weight + (size_t)e * (OUT_ * IN_) + r4);
    ax += g[e] * wv.x; ay += g[e] * wv.y; az += g[e] * wv.z; aw += g[e] * wv.w;
  }
  bf16x4 o;
  o[0] = (__bf16)ax; o[1] = (__bf16)ay; o[2] = (__bf16)az; o[3] = (__bf16)aw;
  *(bf16x4*)(wbf + (size_t)b * (OUT_ * IN_) + r4) = o;
}

// ---------------------------------------------------------------------------
// Kernel 2: mix bias -> [B][OUT] f32
// ---------------------------------------------------------------------------
__global__ __launch_bounds__(256) void moe_mix_b(const float* __restrict__ gates,
                                                 const float* __restrict__ bias,
                                                 float* __restrict__ bmix) {
  const unsigned tid = blockIdx.x * 256u + threadIdx.x;      // B*OUT threads
  const unsigned b = tid >> 9, o = tid & (OUT_ - 1);
  float s = 0.f;
#pragma unroll
  for (int e = 0; e < NE_; ++e) s += gates[b * NE_ + e] * bias[e * OUT_ + o];
  bmix[tid] = s;
}

// ---------------------------------------------------------------------------
// Kernel 2b: convert x (f32) -> xbf (bf16), one pass. 8 elements per thread.
// Removes 8x-redundant in-loop conversions from the compute-bound GEMM.
// ---------------------------------------------------------------------------
__global__ __launch_bounds__(256) void moe_cvt_x(const float* __restrict__ x,
                                                 __bf16* __restrict__ xbf) {
  const size_t e8 = ((size_t)blockIdx.x * 256u + threadIdx.x) * 8;  // B*L*IN/8 thr
  const float4 a0 = *(const float4*)(x + e8);
  const float4 a1 = *(const float4*)(x + e8 + 4);
  bf16x8 o;
  o[0] = (__bf16)a0.x; o[1] = (__bf16)a0.y; o[2] = (__bf16)a0.z; o[3] = (__bf16)a0.w;
  o[4] = (__bf16)a1.x; o[5] = (__bf16)a1.y; o[6] = (__bf16)a1.z; o[7] = (__bf16)a1.w;
  *(bf16x8*)(xbf + e8) = o;
}

// ---------------------------------------------------------------------------
// Kernel 3 (primary): out[b] = xbf[b] @ wbf[b]^T + bmix[b], all-bf16 fragments.
// WG = 256 threads = 8 waves, 4(L) x 2(O); WG tile 128 x 128.
// Wave tile 32(L) x 64(O) = 2x4 accumulators -> 8 WMMAs per k-step.
// Hot loop is pure VMEM (b128 loads) + v_wmma: the MI455X-optimal mix.
// ---------------------------------------------------------------------------
__global__ __launch_bounds__(256) void moe_gemm_bf(const __bf16* __restrict__ xbf,
                                                   const __bf16* __restrict__ wbf,
                                                   const float*  __restrict__ bmix,
                                                   float* __restrict__ out) {
  const int b    = blockIdx.z;
  const int lBlk = blockIdx.y * 128;
  const int oBlk = blockIdx.x * 128;

  const int lane = threadIdx.x & 31;
  const int wave = threadIdx.x >> 5;
  const int wl   = wave & 3;           // 4 waves along L
  const int wo   = wave >> 2;          // 2 waves along OUT
  const int lW   = lBlk + wl * 32;
  const int oW   = oBlk + wo * 64;

  const int half = lane >> 4;          // 0: lanes 0-15, 1: lanes 16-31
  const int lan  = lane & 15;

  const __bf16* xb = xbf + (size_t)b * L_ * IN_;
  const __bf16* wb = wbf + (size_t)b * OUT_ * IN_;

  // Accumulators pre-seeded with mixed bias (depends only on column n = lan).
  f32x8 acc[2][4];
#pragma unroll
  for (int ni = 0; ni < 4; ++ni) {
    const float bv = bmix[b * OUT_ + oW + ni * 16 + lan];
#pragma unroll
    for (int mi = 0; mi < 2; ++mi)
#pragma unroll
      for (int r = 0; r < 8; ++r) acc[mi][ni][r] = bv;
  }

  for (int k0 = 0; k0 < IN_; k0 += 32) {
    // A k order used here (same as B): lane<16 holds K {0..15}, lane>=16 holds
    // K {16..31} of row (lane&15)  -> two contiguous 16B bf16 loads.
    bf16x16 A[2];
#pragma unroll
    for (int mi = 0; mi < 2; ++mi) {
      const __bf16* ar = xb + (size_t)(lW + mi * 16 + lan) * IN_ + k0 + half * 16;
      const bf16x8 lo = *(const bf16x8*)(ar);
      const bf16x8 hi = *(const bf16x8*)(ar + 8);
#pragma unroll
      for (int i = 0; i < 8; ++i) { A[mi][i] = lo[i]; A[mi][i + 8] = hi[i]; }
    }

    // B layout (32x16): lane%16 = column o, lane/16 selects K half ->
    // one contiguous 32B bf16 chunk per lane.
    bf16x16 Bf[4];
#pragma unroll
    for (int ni = 0; ni < 4; ++ni) {
      const __bf16* wr = wb + (size_t)(oW + ni * 16 + lan) * IN_ + k0 + half * 16;
      Bf[ni] = *(const bf16x16*)wr;
    }

#pragma unroll
    for (int mi = 0; mi < 2; ++mi)
#pragma unroll
      for (int ni = 0; ni < 4; ++ni)
        acc[mi][ni] = __builtin_amdgcn_wmma_f32_16x16x32_bf16(
            false, A[mi], false, Bf[ni], (short)0, acc[mi][ni], false, false);
  }

  // D layout: VGPR r is row (half*8 + r), lane%16 is column.
#pragma unroll
  for (int mi = 0; mi < 2; ++mi)
#pragma unroll
    for (int ni = 0; ni < 4; ++ni) {
      const int l0 = lW + mi * 16 + half * 8;
      const int o  = oW + ni * 16 + lan;
      float* op = out + ((size_t)b * L_ + l0) * OUT_ + o;
#pragma unroll
      for (int r = 0; r < 8; ++r) op[(size_t)r * OUT_] = acc[mi][ni][r];
    }
}

// ---------------------------------------------------------------------------
// Kernel 3 (fallback, small workspace): f32 x with in-register bf16 convert.
// ---------------------------------------------------------------------------
__global__ __launch_bounds__(256) void moe_gemm_f32x(const float*  __restrict__ x,
                                                     const __bf16* __restrict__ wbf,
                                                     const float*  __restrict__ bmix,
                                                     float* __restrict__ out) {
  const int b    = blockIdx.z;
  const int lBlk = blockIdx.y * 128;
  const int oBlk = blockIdx.x * 128;

  const int lane = threadIdx.x & 31;
  const int wave = threadIdx.x >> 5;
  const int wl   = wave & 3;
  const int wo   = wave >> 2;
  const int lW   = lBlk + wl * 32;
  const int oW   = oBlk + wo * 64;

  const int half = lane >> 4;
  const int lan  = lane & 15;

  const float*  xb = x   + (size_t)b * L_ * IN_;
  const __bf16* wb = wbf + (size_t)b * OUT_ * IN_;

  f32x8 acc[2][4];
#pragma unroll
  for (int ni = 0; ni < 4; ++ni) {
    const float bv = bmix[b * OUT_ + oW + ni * 16 + lan];
#pragma unroll
    for (int mi = 0; mi < 2; ++mi)
#pragma unroll
      for (int r = 0; r < 8; ++r) acc[mi][ni][r] = bv;
  }

  for (int k0 = 0; k0 < IN_; k0 += 32) {
    bf16x16 A[2];
#pragma unroll
    for (int mi = 0; mi < 2; ++mi) {
      const float* xr = xb + (size_t)(lW + mi * 16 + lan) * IN_ + k0 + half * 16;
      const float4 a0 = *(const float4*)(xr + 0);
      const float4 a1 = *(const float4*)(xr + 4);
      const float4 a2 = *(const float4*)(xr + 8);
      const float4 a3 = *(const float4*)(xr + 12);
      A[mi][0]  = (__bf16)a0.x; A[mi][1]  = (__bf16)a0.y;
      A[mi][2]  = (__bf16)a0.z; A[mi][3]  = (__bf16)a0.w;
      A[mi][4]  = (__bf16)a1.x; A[mi][5]  = (__bf16)a1.y;
      A[mi][6]  = (__bf16)a1.z; A[mi][7]  = (__bf16)a1.w;
      A[mi][8]  = (__bf16)a2.x; A[mi][9]  = (__bf16)a2.y;
      A[mi][10] = (__bf16)a2.z; A[mi][11] = (__bf16)a2.w;
      A[mi][12] = (__bf16)a3.x; A[mi][13] = (__bf16)a3.y;
      A[mi][14] = (__bf16)a3.z; A[mi][15] = (__bf16)a3.w;
    }

    bf16x16 Bf[4];
#pragma unroll
    for (int ni = 0; ni < 4; ++ni) {
      const __bf16* wr = wb + (size_t)(oW + ni * 16 + lan) * IN_ + k0 + half * 16;
      Bf[ni] = *(const bf16x16*)wr;
    }

#pragma unroll
    for (int mi = 0; mi < 2; ++mi)
#pragma unroll
      for (int ni = 0; ni < 4; ++ni)
        acc[mi][ni] = __builtin_amdgcn_wmma_f32_16x16x32_bf16(
            false, A[mi], false, Bf[ni], (short)0, acc[mi][ni], false, false);
  }

#pragma unroll
  for (int mi = 0; mi < 2; ++mi)
#pragma unroll
    for (int ni = 0; ni < 4; ++ni) {
      const int l0 = lW + mi * 16 + half * 8;
      const int o  = oW + ni * 16 + lan;
      float* op = out + ((size_t)b * L_ + l0) * OUT_ + o;
#pragma unroll
      for (int r = 0; r < 8; ++r) op[(size_t)r * OUT_] = acc[mi][ni][r];
    }
}

// ---------------------------------------------------------------------------
extern "C" void kernel_launch(void* const* d_in, const int* in_sizes, int n_in,
                              void* d_out, int out_size, void* d_ws, size_t ws_size,
                              hipStream_t stream) {
  const float* x      = (const float*)d_in[0];  // [32,1024,1024]
  const float* gates  = (const float*)d_in[1];  // [32,8]
  const float* weight = (const float*)d_in[2];  // [8,512,1024]
  const float* bias   = (const float*)d_in[3];  // [8,512]
  float*       out    = (float*)d_out;          // [32,1024,512]

  const size_t xbfBytes = (size_t)B_ * L_ * IN_ * 2;    // 64 MB
  const size_t wbfBytes = (size_t)B_ * OUT_ * IN_ * 2;  // 32 MB
  const size_t bmixBytes = (size_t)B_ * OUT_ * 4;       // 64 KB

  dim3 grid(OUT_ / 128, L_ / 128, B_);

  if (ws_size >= xbfBytes + wbfBytes + bmixBytes) {
    // Full path: pre-convert x once; hot loop is pure b128-load + v_wmma.
    __bf16* xbf  = (__bf16*)d_ws;
    __bf16* wbf  = (__bf16*)((char*)d_ws + xbfBytes);
    float*  bmix = (float*)((char*)d_ws + xbfBytes + wbfBytes);

    moe_cvt_x<<<(B_ * L_ * (IN_ / 8)) / 256, 256, 0, stream>>>(x, xbf);
    moe_mix_w<<<(B_ * OUT_ * IN_ / 4) / 256, 256, 0, stream>>>(gates, weight, wbf);
    moe_mix_b<<<(B_ * OUT_) / 256, 256, 0, stream>>>(gates, bias, bmix);
    moe_gemm_bf<<<grid, 256, 0, stream>>>(xbf, wbf, bmix, out);
  } else {
    // Small-workspace fallback: convert x tiles in registers inside the GEMM.
    __bf16* wbf  = (__bf16*)d_ws;
    float*  bmix = (float*)((char*)d_ws + wbfBytes);

    moe_mix_w<<<(B_ * OUT_ * IN_ / 4) / 256, 256, 0, stream>>>(gates, weight, wbf);
    moe_mix_b<<<(B_ * OUT_) / 256, 256, 0, stream>>>(gates, bias, bmix);
    moe_gemm_f32x<<<grid, 256, 0, stream>>>(x, wbf, bmix, out);
  }
  (void)in_sizes; (void)n_in; (void)out_size;
}
